// Bi_path_attention_21260088115969
// MI455X (gfx1250) — compile-verified, matrix operand order checked by reading, not run on hardware
//
#include <hip/hip_runtime.h>

// CDNA5 / gfx1250 WMMA types
typedef __attribute__((ext_vector_type(16))) __bf16 v16bf;
typedef __attribute__((ext_vector_type(8)))  float  v8f;

#define WMMA_BF16(a, b, c) \
  __builtin_amdgcn_wmma_f32_16x16x32_bf16(false, (a), false, (b), (short)0, (c), false, false)

static __device__ __forceinline__ unsigned short f2bf(float f) {
  unsigned u = __float_as_uint(f);
  u += 0x7fffu + ((u >> 16) & 1u);            // round-to-nearest-even
  return (unsigned short)(u >> 16);
}
static __device__ __forceinline__ float bf2f(unsigned short h) {
  return __uint_as_float(((unsigned)h) << 16);
}
static __device__ __forceinline__ int imin(int a, int b) { return a < b ? a : b; }
static __device__ __forceinline__ int imax(int a, int b) { return a > b ? a : b; }

// ---------------------------------------------------------------------------
// Fragment swizzle convention (assumed CDNA5 16-bit WMMA layout, ISA 7.12.2):
//  A (16xK): lane = M + 16*kHalf, 16 bf16/lane per 32-K tile, contiguous.
//  B (Kx16): lane = N + 16*kHalf, same K packing (lanes hold N, VGPRs hold K).
//  For K element kl in [0,32): e = (kl&7) | ((kl&16)>>1), half = (kl>>3)&1.
// Both A and B sides use the same convention, built by the same swizzler.
// ---------------------------------------------------------------------------

// K0: f32 weights -> bf16, pre-swizzled into B-fragment tiles.
// dst[((NT*Ktiles + kt)*32 + lane)*16 + e], NT = n/16, Ktiles = K/32.
__global__ void k0_weight_swz(const float* __restrict__ Wsrc,
                              unsigned short* __restrict__ Wdst, int K, int N) {
  int idx = blockIdx.x * blockDim.x + threadIdx.x;
  int total = K * N;
  if (idx >= total) return;
  int k = idx / N, n = idx - k * N;
  int NT = n >> 4, nl = n & 15, kt = k >> 5, kl = k & 31;
  int e  = (kl & 7) | ((kl & 16) >> 1);
  int ln = nl + (((kl >> 3) & 1) << 4);
  int Kt = K >> 5;
  Wdst[(((size_t)NT * Kt + kt) * 32 + ln) * 16 + e] = f2bf(Wsrc[(size_t)k * N + n]);
}

// K1: AvgPool2d(2) on x viewed as (B,C,128,128) -> pooled (B,C,64,64)
__global__ void k1_avgpool(const float* __restrict__ x, float* __restrict__ p) {
  int idx = blockIdx.x * blockDim.x + threadIdx.x;
  if (idx >= 4 * 512 * 64 * 64) return;
  int xw = idx & 63, yh = (idx >> 6) & 63, bc = idx >> 12;  // bc = b*512 + c
  const float* s = x + (size_t)bc * 16384 + (yh * 2) * 128 + xw * 2;
  p[idx] = 0.25f * (s[0] + s[1] + s[128] + s[129]);
}

// K2: fused QKV GEMM (bf16 WMMA) + 2x2 windowed attention.
// Block: 256 threads = 8 waves, 32 tokens = 8 windows.
// MODE 0 (global path): output o as bf16 in A-fragment swizzle (feeds K3).
// MODE 1 (local path):  output o as f32, token-major (B,h,w,C) flat.
template <int MODE>
__global__ __launch_bounds__(256)
void k2_window_attn(const float* __restrict__ src,
                    const unsigned short* __restrict__ Wq,
                    const float* __restrict__ bq,
                    void* __restrict__ outp, int Hs) {
  const int wgc = Hs >> 1;
  const int wpi = wgc * wgc;  // windows per image (H==W)
  __shared__ __align__(64) unsigned short lA[2 * 16 * 512];  // 32KB, A-frag swizzled
  __shared__ __align__(64) unsigned short lQKV[32 * 192];    // 12KB, per-head q|k|v
  __shared__ float lAttn[8][4][4];
  __shared__ int   lTok[32];
  const int t = threadIdx.x;
  const int bid = blockIdx.x;

  if (t < 32) {  // token flat offsets (window-grouped order, matches K3)
    int m = t, j = m >> 2;
    int wi = bid * 8 + j;
    int b  = wi / wpi; int r = wi - b * wpi;
    int hy = r / wgc;  int wx = r - hy * wgc;
    int hh = 2 * hy + ((m >> 1) & 1);
    int ww = 2 * wx + (m & 1);
    lTok[m] = ((b * Hs + hh) * Hs + ww) * 512;
  }
  __syncthreads();

  // Load 32 token rows (coalesced), convert bf16, store A-fragment swizzled.
  for (int idx = t; idx < 32 * 512; idx += 256) {
    int m = idx >> 9, k = idx & 511;
    float v = src[(size_t)lTok[m] + k];
    int kt = k >> 5, kl = k & 31;
    int e  = (kl & 7) | ((kl & 16) >> 1);
    int ln = (m & 15) + (((kl >> 3) & 1) << 4);
    lA[(m >> 4) * 8192 + kt * 512 + ln * 16 + e] = f2bf(v);
  }
  __syncthreads();

  const int wv = t >> 5, lane = t & 31;
  const int nl = lane & 15;
  const int mt = wv >> 2, wn = wv & 3;
  const int rB = mt * 16 + ((lane >> 4) << 3);
  const unsigned short* aB = &lA[mt * 8192 + lane * 16];

  for (int head = 0; head < 8; ++head) {
    // 12 column tiles of this head's q|k|v: tile tt -> qkv col tile p*32+head*4+sub
    int NT[3], colB[3];
#pragma unroll
    for (int s = 0; s < 3; ++s) {
      int tt = wn * 3 + s;
      int p = tt >> 2, sub = tt & 3;
      NT[s]   = p * 32 + head * 4 + sub;
      colB[s] = p * 64 + sub * 16 + nl;  // column inside lQKV row (q:0 k:64 v:128)
    }
    const unsigned short* bp0 = Wq + (size_t)NT[0] * 8192 + lane * 16;
    const unsigned short* bp1 = Wq + (size_t)NT[1] * 8192 + lane * 16;
    const unsigned short* bp2 = Wq + (size_t)NT[2] * 8192 + lane * 16;
    v8f acc0 = {}, acc1 = {}, acc2 = {};
#pragma unroll 4
    for (int kt = 0; kt < 16; ++kt) {
      v16bf a  = *reinterpret_cast<const v16bf*>(aB + kt * 512);
      v16bf b0 = *reinterpret_cast<const v16bf*>(bp0 + kt * 512);
      v16bf b1 = *reinterpret_cast<const v16bf*>(bp1 + kt * 512);
      v16bf b2 = *reinterpret_cast<const v16bf*>(bp2 + kt * 512);
      __builtin_prefetch(bp0 + kt * 512 + 512, 0, 1);  // global_prefetch_b8
      acc0 = WMMA_BF16(a, b0, acc0);
      acc1 = WMMA_BF16(a, b1, acc1);
      acc2 = WMMA_BF16(a, b2, acc2);
    }
    float bias0 = bq[NT[0] * 16 + nl];
    float bias1 = bq[NT[1] * 16 + nl];
    float bias2 = bq[NT[2] * 16 + nl];
#pragma unroll
    for (int r = 0; r < 8; ++r) {
      int row = (rB + r) * 192;
      lQKV[row + colB[0]] = f2bf(acc0[r] + bias0);
      lQKV[row + colB[1]] = f2bf(acc1[r] + bias1);
      lQKV[row + colB[2]] = f2bf(acc2[r] + bias2);
    }
    __syncthreads();

    // Scores + softmax: one thread per (window j, query i); n = 4 tokens.
    if (t < 32) {
      int j = t >> 2, i = t & 3;
      const unsigned short* q = &lQKV[(j * 4 + i) * 192];
      float s[4]; float mx = -3.4e38f;
#pragma unroll
      for (int jj = 0; jj < 4; ++jj) {
        const unsigned short* kr = &lQKV[(j * 4 + jj) * 192 + 64];
        float d = 0.f;
        for (int kk = 0; kk < 64; ++kk) d += bf2f(q[kk]) * bf2f(kr[kk]);
        s[jj] = d * 0.125f;                 // scale = 64^-0.5
        mx = fmaxf(mx, s[jj]);
      }
      float sum = 0.f;
#pragma unroll
      for (int jj = 0; jj < 4; ++jj) { s[jj] = __expf(s[jj] - mx); sum += s[jj]; }
      float inv = 1.f / sum;
#pragma unroll
      for (int jj = 0; jj < 4; ++jj) lAttn[j][i][jj] = s[jj] * inv;
    }
    __syncthreads();

    // o = attn @ v ; 256 threads: (window j, query i, 8-dim chunk dc)
    {
      int j = t >> 5, i = (t >> 3) & 3, dc = t & 7;
      float a0 = lAttn[j][i][0], a1 = lAttn[j][i][1];
      float a2 = lAttn[j][i][2], a3 = lAttn[j][i][3];
      const unsigned short* v0 = &lQKV[(j * 4) * 192 + 128 + dc * 8];
      const unsigned short* v1 = v0 + 192;
      const unsigned short* v2 = v0 + 384;
      const unsigned short* v3 = v0 + 576;
      float ov[8];
#pragma unroll
      for (int d = 0; d < 8; ++d)
        ov[d] = a0 * bf2f(v0[d]) + a1 * bf2f(v1[d]) + a2 * bf2f(v2[d]) + a3 * bf2f(v3[d]);
      int m  = j * 4 + i;
      int cb = head * 64 + dc * 8;
      if constexpr (MODE == 0) {
        union { uint4 u; unsigned short h[8]; } pk;
#pragma unroll
        for (int d = 0; d < 8; ++d) pk.h[d] = f2bf(ov[d]);
        int kt = cb >> 5, kl = cb & 31;
        int e  = (kl & 7) | ((kl & 16) >> 1);
        int ln = (m & 15) + (((kl >> 3) & 1) << 4);
        unsigned short* dst = (unsigned short*)outp + (size_t)bid * 16384 +
                              (m >> 4) * 8192 + kt * 512 + ln * 16 + e;
        *reinterpret_cast<uint4*>(dst) = pk.u;
      } else {
        float* dst = (float*)outp + (size_t)lTok[m] + cb;
        float4* d4 = reinterpret_cast<float4*>(dst);
        d4[0] = make_float4(ov[0], ov[1], ov[2], ov[3]);
        d4[1] = make_float4(ov[4], ov[5], ov[6], ov[7]);
      }
    }
    __syncthreads();  // protect lQKV before next head
  }
}

// K2.5: local smoothing (reflect-pad avg in x and y) + bilinear x2 upsample.
// Ol is the token-major local attention output; read via its raw (B,C,64,64) view.
// Lup produced in (B,C,128,128) layout.
__global__ void k25_smooth_upsample(const float* __restrict__ Ol,
                                    float* __restrict__ Lup) {
  int idx = blockIdx.x * blockDim.x + threadIdx.x;
  if (idx >= 4 * 512 * 128 * 128) return;
  int w = idx & 127, h = (idx >> 7) & 127;
  int c = (idx >> 14) & 511, b = idx >> 23;
  const float* base = Ol + ((size_t)b * 512 + c) * 4096;  // raw-reshape view
  float sy = 0.5f * h - 0.25f, sx = 0.5f * w - 0.25f;     // half-pixel centers
  int y0 = (int)floorf(sy), x0 = (int)floorf(sx);
  float fy = sy - (float)y0, fx = sx - (float)x0;
  int y0c = imin(imax(y0, 0), 63), y1c = imin(imax(y0 + 1, 0), 63);
  int x0c = imin(imax(x0, 0), 63), x1c = imin(imax(x0 + 1, 0), 63);
  auto S = [&](int y, int x) -> float {
    int ym = (y == 0) ? 1 : y - 1;      // reflect-pad top
    int xp = (x == 63) ? 62 : x + 1;    // reflect-pad right
    float lv = base[y * 64 + x];
    return 0.5f * (base[ym * 64 + x] + lv) + 0.5f * (lv + base[y * 64 + xp]);
  };
  float v = (1.f - fy) * ((1.f - fx) * S(y0c, x0c) + fx * S(y0c, x1c))
          +        fy  * ((1.f - fx) * S(y1c, x0c) + fx * S(y1c, x1c));
  Lup[idx] = v;
}

// K3: proj GEMM (o @ Wproj + bproj) + add local field + scatter to (B,C,H,W).
// Raw reshape of Lup (B,C,H,W)->(B,H,W,C) makes the local add token-major.
__global__ __launch_bounds__(256)
void k3_proj_combine(const unsigned short* __restrict__ Og,
                     const unsigned short* __restrict__ Wp,
                     const float* __restrict__ bp,
                     const float* __restrict__ Lup,
                     float* __restrict__ out) {
  __shared__ int lB[32], lH[32], lW[32];
  const int t = threadIdx.x, bid = blockIdx.x;
  if (t < 32) {  // same window-grouped token mapping as K2<0>
    int m = t, j = m >> 2;
    int wi = bid * 8 + j;
    int b = wi >> 12; int r = wi & 4095;  // wpi = 64*64
    int hy = r >> 6, wx = r & 63;
    lB[m] = b; lH[m] = 2 * hy + ((m >> 1) & 1); lW[m] = 2 * wx + (m & 1);
  }
  __syncthreads();
  const int wv = t >> 5, lane = t & 31, nl = lane & 15;
  const int mt = wv >> 2, wn = wv & 3;
  const unsigned short* aB = Og + (size_t)bid * 16384 + mt * 8192 + lane * 16;
  v8f acc[8];
#pragma unroll
  for (int s = 0; s < 8; ++s) acc[s] = (v8f){};
#pragma unroll 2
  for (int kt = 0; kt < 16; ++kt) {
    v16bf a = *reinterpret_cast<const v16bf*>(aB + kt * 512);
#pragma unroll
    for (int s = 0; s < 8; ++s) {
      int NT = wn * 8 + s;
      const unsigned short* wptr = Wp + (size_t)NT * 8192 + kt * 512 + lane * 16;
      v16bf b = *reinterpret_cast<const v16bf*>(wptr);
      acc[s] = WMMA_BF16(a, b, acc[s]);
    }
    __builtin_prefetch(Wp + (size_t)(wn * 8) * 8192 + kt * 512 + lane * 16 + 512, 0, 1);
  }
#pragma unroll
  for (int s = 0; s < 8; ++s) {
    int c = (wn * 8 + s) * 16 + nl;
    float bias = bp[c];
#pragma unroll
    for (int r = 0; r < 8; ++r) {
      int m = mt * 16 + ((lane >> 4) << 3) + r;
      size_t tok  = ((size_t)lB[m] * 128 + lH[m]) * 128 + lW[m];
      size_t oidx = (((size_t)lB[m] * 512 + c) * 128 + lH[m]) * 128 + lW[m];
      out[oidx] = acc[s][r] + bias + Lup[tok * 512 + c];
    }
  }
}

extern "C" void kernel_launch(void* const* d_in, const int* in_sizes, int n_in,
                              void* d_out, int out_size, void* d_ws, size_t ws_size,
                              hipStream_t stream) {
  (void)in_sizes; (void)n_in; (void)out_size; (void)ws_size;
  const float* x     = (const float*)d_in[0];
  const float* Wqkv  = (const float*)d_in[1];
  const float* bqkv  = (const float*)d_in[2];
  const float* Wproj = (const float*)d_in[3];
  const float* bproj = (const float*)d_in[4];
  float* out = (float*)d_out;
  char* ws = (char*)d_ws;
  const size_t MB = 1ull << 20;
  // Scratch layout (~260 MB total):
  unsigned short* wsWq   = (unsigned short*)(ws);             // 1.5 MB  swizzled Wqkv bf16
  unsigned short* wsWp   = (unsigned short*)(ws + 2 * MB);    // 0.5 MB  swizzled Wproj bf16
  float*          wsPool = (float*)(ws + 3 * MB);             // 32 MB   pooled (B,C,64,64)
  unsigned short* wsOg   = (unsigned short*)(ws + 35 * MB);   // 64 MB   global o, bf16 A-swizzled
  float*          wsOl   = (float*)(ws + 99 * MB);            // 32 MB   local o, f32 token-major
  float*          wsLup  = (float*)(ws + 131 * MB);           // 128 MB  upsampled local field

  k0_weight_swz<<<(512 * 1536 + 255) / 256, 256, 0, stream>>>(Wqkv, wsWq, 512, 1536);
  k0_weight_swz<<<(512 * 512 + 255) / 256, 256, 0, stream>>>(Wproj, wsWp, 512, 512);
  k1_avgpool<<<(4 * 512 * 64 * 64 + 255) / 256, 256, 0, stream>>>(x, wsPool);
  k2_window_attn<0><<<2048, 256, 0, stream>>>(x, wsWq, bqkv, (void*)wsOg, 128);
  k2_window_attn<1><<<512, 256, 0, stream>>>(wsPool, wsWq, bqkv, (void*)wsOl, 64);
  k25_smooth_upsample<<<(4 * 512 * 128 * 128 + 255) / 256, 256, 0, stream>>>(wsOl, wsLup);
  k3_proj_combine<<<2048, 256, 0, stream>>>(wsOg, wsWp, bproj, wsLup, out);
}
